// KEPCE_GAT_51419348467712
// MI455X (gfx1250) — matrix operand order repaired and checked by reading, not compiled, not run on hardware
//
#include <hip/hip_runtime.h>
#include <hip/hip_bf16.h>
#include <math.h>

typedef __attribute__((ext_vector_type(16))) _Float16 v16h;
typedef __attribute__((ext_vector_type(8)))  float    v8f;

#define NEG_SLOPE 0.2f
#define GAT_H 4

// ---------------------------------------------------------------------------
// small helpers
// ---------------------------------------------------------------------------
__device__ __forceinline__ unsigned kepce_fflip(float f) {
    unsigned u = __float_as_uint(f);
    return (u & 0x80000000u) ? ~u : (u | 0x80000000u);
}
__device__ __forceinline__ float kepce_funflip(unsigned u) {
    return (u & 0x80000000u) ? __uint_as_float(u & 0x7fffffffu)
                             : __uint_as_float(~u);
}

__global__ void kepce_fill(float* __restrict__ p, long n, float v) {
    long i = (long)blockIdx.x * blockDim.x + threadIdx.x;
    if (i < n) p[i] = v;
}

// out[i*M+m] = relu(b[m] + sum_k in[i*K+k] * W[k*M+m])
__global__ void kepce_linear_relu(const float* __restrict__ in,
                                  const float* __restrict__ W,
                                  const float* __restrict__ b,
                                  float* __restrict__ out,
                                  long N, int K, int M) {
    long t = (long)blockIdx.x * blockDim.x + threadIdx.x;
    if (t >= N * (long)M) return;
    long i = t / M;
    int  m = (int)(t % M);
    float acc = b[m];
    const float* row = in + i * K;
    for (int k = 0; k < K; ++k) acc += row[k] * W[(long)k * M + m];
    out[t] = acc > 0.0f ? acc : 0.0f;
}

// xl = in@Wl + bl ; xr = in@Wr + br   (no activation)
__global__ void kepce_dual_linear(const float* __restrict__ in,
                                  const float* __restrict__ Wl,
                                  const float* __restrict__ bl,
                                  const float* __restrict__ Wr,
                                  const float* __restrict__ br,
                                  float* __restrict__ xl,
                                  float* __restrict__ xr,
                                  long N, int K, int M) {
    long t = (long)blockIdx.x * blockDim.x + threadIdx.x;
    if (t >= N * (long)M) return;
    long i = t / M;
    int  m = (int)(t % M);
    float al = bl[m], ar = br[m];
    const float* row = in + i * K;
    for (int k = 0; k < K; ++k) {
        float v = row[k];
        al += v * Wl[(long)k * M + m];
        ar += v * Wr[(long)k * M + m];
    }
    xl[t] = al;
    xr[t] = ar;
}

// degree + edge-attr sum per dst node
__global__ void kepce_degree(const long long* __restrict__ ei,
                             const float* __restrict__ ew,
                             const float* __restrict__ ce,
                             float* __restrict__ cnt,
                             float* __restrict__ lsum,   // [N,2]
                             long E) {
    long e = (long)blockIdx.x * blockDim.x + threadIdx.x;
    if (e >= E) return;
    long d = (long)ei[E + e];
    atomicAdd(&cnt[d], 1.0f);
    atomicAdd(&lsum[d * 2 + 0], ew[e]);
    atomicAdd(&lsum[d * 2 + 1], ce[e]);
}

__global__ void kepce_loopnorm(float* __restrict__ lat,
                               const float* __restrict__ cnt, long N) {
    long i = (long)blockIdx.x * blockDim.x + threadIdx.x;
    if (i >= N) return;
    float c = cnt[i];
    if (c < 1.0f) c = 1.0f;
    lat[i * 2 + 0] /= c;
    lat[i * 2 + 1] /= c;
}

// ---------------------------------------------------------------------------
// GATv2 edge passes (edges 0..E-1 real, E..E+N-1 self loops)
// ---------------------------------------------------------------------------
__global__ void kepce_edge_alpha(const float* __restrict__ xl,
                                 const float* __restrict__ xr,
                                 const long long* __restrict__ ei,
                                 const float* __restrict__ ew,
                                 const float* __restrict__ ce,
                                 const float* __restrict__ lat,   // [N,2]
                                 const float* __restrict__ We,    // [2,M]
                                 const float* __restrict__ att,   // [H,C]
                                 float* __restrict__ alpha,       // [E+N,H]
                                 unsigned* __restrict__ amax,     // [N,H] flipped
                                 long E, long N, int C, int M) {
    long e = (long)blockIdx.x * blockDim.x + threadIdx.x;
    if (e >= E + N) return;
    long s, d;
    float ea0, ea1;
    if (e < E) {
        s = (long)ei[e];
        d = (long)ei[E + e];
        ea0 = ew[e];
        ea1 = ce[e];
    } else {
        s = d = e - E;
        ea0 = lat[s * 2 + 0];
        ea1 = lat[s * 2 + 1];
    }
    const float* xls = xl + s * M;
    const float* xrd = xr + d * M;
    for (int h = 0; h < GAT_H; ++h) {
        float acc = 0.0f;
        for (int c = 0; c < C; ++c) {
            int j = h * C + c;
            float g = xls[j] + xrd[j] + ea0 * We[j] + ea1 * We[M + j];
            g = g > 0.0f ? g : NEG_SLOPE * g;
            acc += g * att[j];
        }
        alpha[e * GAT_H + h] = acc;
        atomicMax(&amax[d * GAT_H + h], kepce_fflip(acc));
    }
}

__global__ void kepce_edge_expsum(float* __restrict__ alpha,
                                  const unsigned* __restrict__ amax,
                                  float* __restrict__ den,
                                  const long long* __restrict__ ei,
                                  long E, long N) {
    long e = (long)blockIdx.x * blockDim.x + threadIdx.x;
    if (e >= E + N) return;
    long d = (e < E) ? (long)ei[E + e] : (e - E);
    for (int h = 0; h < GAT_H; ++h) {
        float mx = kepce_funflip(amax[d * GAT_H + h]);
        float ex = expf(alpha[e * GAT_H + h] - mx);
        alpha[e * GAT_H + h] = ex;
        atomicAdd(&den[d * GAT_H + h], ex);
    }
}

// one thread per (edge, head); C atomic adds each
__global__ void kepce_edge_aggr(const float* __restrict__ alpha,
                                const float* __restrict__ den,
                                const float* __restrict__ xl,
                                const long long* __restrict__ ei,
                                float* __restrict__ out,   // [N,M] zeroed
                                long E, long N, int C, int M) {
    long t = (long)blockIdx.x * blockDim.x + threadIdx.x;
    if (t >= (E + N) * GAT_H) return;
    int  h = (int)(t & (GAT_H - 1));
    long e = t >> 2;
    long s, d;
    if (e < E) {
        s = (long)ei[e];
        d = (long)ei[E + e];
    } else {
        s = d = e - E;
    }
    float a = alpha[e * GAT_H + h] / den[d * GAT_H + h];
    const float* xls = xl + s * M + h * C;
    float* dst = out + d * M + h * C;
    for (int c = 0; c < C; ++c) atomicAdd(&dst[c], xls[c] * a);
}

__global__ void kepce_bias_relu(float* __restrict__ p,
                                const float* __restrict__ bias,
                                long N, int M) {
    long t = (long)blockIdx.x * blockDim.x + threadIdx.x;
    if (t >= N * (long)M) return;
    float v = p[t] + bias[t % M];
    p[t] = v > 0.0f ? v : 0.0f;
}

// ---------------------------------------------------------------------------
// FC 32->32 (+bias, relu) via WMMA; emits f16 output (feeds the edge MLP).
// One wave per 16-node tile; K=32 (one k-step), N=32 (two B fragments).
// A-layout loads come straight from global (rows are 32 contiguous f32).
// No LDS, no barriers: waves iterate independently.
// ---------------------------------------------------------------------------
__global__ void __launch_bounds__(128)
kepce_fc_wmma(const float* __restrict__ hin,   // [N,32]
              const float* __restrict__ W,     // [32,32]
              const float* __restrict__ bias,  // [32]
              _Float16* __restrict__ out16,    // [N,32] f16
              long N) {
    const int lane = threadIdx.x & 31;
    const int wv   = threadIdx.x >> 5;
    const long tiles = (N + 15) >> 4;

    // B fragments (invariant): lane holds column n=(l&15); halves j enumerate
    // K = (l>>4)*16 + j.
    v16h B0 = {}, B1 = {};
    {
        const int nB = lane & 15;
        const int kg = (lane >> 4) << 4;
#pragma unroll
        for (int j = 0; j < 16; ++j) {
            B0[j] = (_Float16)W[(long)(kg + j) * 32 + nB];
            B1[j] = (_Float16)W[(long)(kg + j) * 32 + 16 + nB];
        }
    }
    const int n  = lane & 15;
    const int mb = (lane >> 4) << 3;
    const float bn0 = bias[n], bn1 = bias[n + 16];

    for (long tile = (long)blockIdx.x * 4 + wv; tile < tiles;
         tile += (long)gridDim.x * 4) {
        // A fragment: lane l -> row m=(l&15); K = (l>>4)*16 + j.
        const int m  = lane & 15;
        const int kb = (lane >> 4) << 4;
        long r = tile * 16 + m;
        if (r >= N) r = N - 1;               // clamp loads; stores guarded
        const float* row = hin + r * 32;
        v16h a = {};
#pragma unroll
        for (int j = 0; j < 8; ++j) {
            a[j]     = (_Float16)row[kb + j];
            a[j + 8] = (_Float16)row[kb + 8 + j];
        }
        v8f c0 = {}, c1 = {};
        c0 = __builtin_amdgcn_wmma_f32_16x16x32_f16(false, a, false, B0,
                                                    (short)0, c0, false, false);
        c1 = __builtin_amdgcn_wmma_f32_16x16x32_f16(false, a, false, B1,
                                                    (short)0, c1, false, false);
        // C layout: vgpr i -> row mb+i, col n (c0) / n+16 (c1)
#pragma unroll
        for (int i = 0; i < 8; ++i) {
            long rr = tile * 16 + mb + i;
            if (rr < N) {
                float v0 = c0[i] + bn0;
                float v1 = c1[i] + bn1;
                out16[rr * 32 + n]      = (_Float16)(v0 > 0.0f ? v0 : 0.0f);
                out16[rr * 32 + n + 16] = (_Float16)(v1 > 0.0f ? v1 : 0.0f);
            }
        }
    }
}

// ---------------------------------------------------------------------------
// Edge MLP: [E,66] @ w_e1[66,32] (+b_e1) @ w_e2[32,2] (+b_e2) via WMMA f16.
// 16 edges per wave tile; K padded 66 -> 96 (3 k-steps of 32); N=32 as 2 tiles.
// Second GEMM (32->2) fused via register butterfly reduction (no LDS tail).
// ---------------------------------------------------------------------------
__global__ void __launch_bounds__(128)
kepce_edge_mlp_wmma(const _Float16* __restrict__ hfc,   // [N,32] f16
                    const long long* __restrict__ ei,
                    const float* __restrict__ ew,
                    const float* __restrict__ ce,
                    const float* __restrict__ w_e1,     // [66,32]
                    const float* __restrict__ b_e1,     // [32]
                    const float* __restrict__ w_e2,     // [32,2]
                    const float* __restrict__ b_e2,     // [2]
                    float* __restrict__ out,            // [E,2]
                    long E) {
    __shared__ _Float16 smA[4][16][104];   // 96 used + pad (bank spread)

    const int lane = threadIdx.x & 31;
    const int wv   = threadIdx.x >> 5;
    const long tilesTotal = (E + 15) >> 4;

    // Loop-invariant B fragments (B layout used here: lane = column n=(l&15),
    // halves j enumerate K = (l>>4)*16 + j).
    v16h Bf[3][2];
    {
        const int nB = lane & 15;
        const int kg = (lane >> 4) << 4;
        for (int kt = 0; kt < 3; ++kt)
            for (int nt = 0; nt < 2; ++nt) {
                v16h b = {};
#pragma unroll
                for (int j = 0; j < 16; ++j) {
                    int k = kt * 32 + kg + j;
                    float w = (k < 66) ? w_e1[(long)k * 32 + nt * 16 + nB] : 0.0f;
                    b[j] = (_Float16)w;
                }
                Bf[kt][nt] = b;
            }
    }

    // Hoisted per-lane second-GEMM coefficients (kills in-loop s_loads).
    const int n   = lane & 15;
    const int mb  = (lane >> 4) << 3;
    const float be1a = b_e1[n], be1b = b_e1[n + 16];
    const float w2a0 = w_e2[n * 2 + 0],        w2a1 = w_e2[n * 2 + 1];
    const float w2b0 = w_e2[(n + 16) * 2 + 0], w2b1 = w_e2[(n + 16) * 2 + 1];
    const float be20 = b_e2[0], be21 = b_e2[1];

    for (long t0 = (long)blockIdx.x * 4; t0 < tilesTotal; t0 += (long)gridDim.x * 4) {
        const long tile = t0 + wv;
        const bool act = tile < tilesTotal;

        // ---- stage 16x96 f16 feature tile into LDS (2 lanes per row) ----
        {
            const int row  = lane >> 1;
            const int half = lane & 1;
            long e = tile * 16 + row;
            const bool ev = act && (e < E);
            long es = ev ? e : 0;
            long s = (long)ei[es];
            long d = (long)ei[E + es];
            const _Float16* hs = hfc + s * 32;
            const _Float16* hd = hfc + d * 32;
            const int k0 = half * 48;
#pragma unroll 4
            for (int j = 0; j < 48; ++j) {
                int k = k0 + j;
                _Float16 v = (_Float16)0.0f;
                if (ev) {
                    if (k < 2)       v = (_Float16)((k == 0) ? ew[es] : ce[es]);
                    else if (k < 34) v = hs[k - 2];
                    else if (k < 66) v = hd[k - 34];
                }
                smA[wv][row][k] = v;
            }
        }
        __syncthreads();

        // ---- A fragments + WMMA (EXEC all-ones here) ----
        v8f c0 = {}, c1 = {};
        {
            const int m  = lane & 15;
            const int kb = (lane >> 4) << 4;   // K = (l>>4)*16 + j, matches B
#pragma unroll
            for (int kt = 0; kt < 3; ++kt) {
                v16h a = {};
#pragma unroll
                for (int j = 0; j < 8; ++j) {
                    a[j]     = smA[wv][m][kt * 32 + kb + j];
                    a[j + 8] = smA[wv][m][kt * 32 + kb + 8 + j];
                }
                c0 = __builtin_amdgcn_wmma_f32_16x16x32_f16(
                        false, a, false, Bf[kt][0], (short)0, c0, false, false);
                c1 = __builtin_amdgcn_wmma_f32_16x16x32_f16(
                        false, a, false, Bf[kt][1], (short)0, c1, false, false);
            }
        }
        __syncthreads();   // protect smA before next iteration's staging

        // ---- fused 32->2 GEMM: per-lane partials + 16-lane butterfly ----
        // lane holds e1 columns n (c0) and n+16 (c1) for rows mb..mb+7
        float t0a[8], t1a[8];
#pragma unroll
        for (int i = 0; i < 8; ++i) {
            float ea = c0[i] + be1a;
            float eb = c1[i] + be1b;
            t0a[i] = ea * w2a0 + eb * w2b0;
            t1a[i] = ea * w2a1 + eb * w2b1;
        }
#pragma unroll
        for (int msk = 1; msk < 16; msk <<= 1) {
#pragma unroll
            for (int i = 0; i < 8; ++i) {
                t0a[i] += __shfl_xor(t0a[i], msk, 32);
                t1a[i] += __shfl_xor(t1a[i], msk, 32);
            }
        }
        // writer: lane (l&15) = 2*i + o  ->  out[(tile*16+mb+i)*2 + o]
        {
            const int isel = (lane & 15) >> 1;
            const int osel = lane & 1;
            float v = 0.0f;
#pragma unroll
            for (int i = 0; i < 8; ++i)
                if (isel == i) v = osel ? t1a[i] : t0a[i];
            v += osel ? be21 : be20;
            long e = tile * 16 + mb + isel;
            if (act && e < E) out[e * 2 + osel] = v;
        }
    }
}

// ---------------------------------------------------------------------------
// host launcher
// ---------------------------------------------------------------------------
static inline int kepce_cdiv(long a, long b) { return (int)((a + b - 1) / b); }

extern "C" void kernel_launch(void* const* d_in, const int* in_sizes, int n_in,
                              void* d_out, int out_size, void* d_ws, size_t ws_size,
                              hipStream_t stream) {
    const float*     x      = (const float*)d_in[0];
    const long long* ei     = (const long long*)d_in[1];   // int64 per reference
    const float*     ew     = (const float*)d_in[2];
    const float*     ce     = (const float*)d_in[3];
    const float*     w_init = (const float*)d_in[4];
    const float*     b_init = (const float*)d_in[5];
    const float*     w1l    = (const float*)d_in[6];
    const float*     b1l    = (const float*)d_in[7];
    const float*     w1r    = (const float*)d_in[8];
    const float*     b1r    = (const float*)d_in[9];
    const float*     w1e    = (const float*)d_in[10];
    const float*     att1   = (const float*)d_in[11];
    const float*     bias1  = (const float*)d_in[12];
    const float*     w2l    = (const float*)d_in[13];
    const float*     b2l    = (const float*)d_in[14];
    const float*     w2r    = (const float*)d_in[15];
    const float*     b2r    = (const float*)d_in[16];
    const float*     w2e    = (const float*)d_in[17];
    const float*     att2   = (const float*)d_in[18];
    const float*     bias2  = (const float*)d_in[19];
    const float*     w_fc   = (const float*)d_in[20];
    const float*     b_fc   = (const float*)d_in[21];
    const float*     w_e1   = (const float*)d_in[22];
    const float*     b_e1   = (const float*)d_in[23];
    const float*     w_e2   = (const float*)d_in[24];
    const float*     b_e2   = (const float*)d_in[25];

    const long N = in_sizes[0] / 5;
    const long E = in_sizes[2];
    const long EN = E + N;

    float* ws = (float*)d_ws;
    size_t o = 0;
    float* hA    = ws + o; o += (size_t)N * 32;    // h0 / h2
    float* hB    = ws + o; o += (size_t)N * 32;    // h1 / f16 h_fc
    float* xl    = ws + o; o += (size_t)N * 32;
    float* xr    = ws + o; o += (size_t)N * 32;
    float* cnt   = ws + o; o += (size_t)N;
    float* lat   = ws + o; o += (size_t)N * 2;     // loop edge-attr [N,2]
    unsigned* amax = (unsigned*)(ws + o); o += (size_t)N * GAT_H;
    float* den   = ws + o; o += (size_t)N * GAT_H;
    float* alpha = ws + o; o += (size_t)EN * GAT_H;

    const int B = 256;

    // --- degree / self-loop attr (shared by both GAT layers) ---
    kepce_fill<<<kepce_cdiv(N, B), B, 0, stream>>>(cnt, N, 0.0f);
    kepce_fill<<<kepce_cdiv(N * 2, B), B, 0, stream>>>(lat, N * 2, 0.0f);
    kepce_degree<<<kepce_cdiv(E, B), B, 0, stream>>>(ei, ew, ce, cnt, lat, E);
    kepce_loopnorm<<<kepce_cdiv(N, B), B, 0, stream>>>(lat, cnt, N);

    // --- init node linear 5->8 (relu) into hA ---
    kepce_linear_relu<<<kepce_cdiv(N * 8, B), B, 0, stream>>>(x, w_init, b_init, hA, N, 5, 8);

    // ================= GAT layer 1: in 8, M=16, C=4 -> hB =================
    kepce_dual_linear<<<kepce_cdiv(N * 16, B), B, 0, stream>>>(hA, w1l, b1l, w1r, b1r, xl, xr, N, 8, 16);
    kepce_fill<<<kepce_cdiv(N * GAT_H, B), B, 0, stream>>>((float*)amax, N * GAT_H, 0.0f); // flip-coded -inf
    kepce_fill<<<kepce_cdiv(N * GAT_H, B), B, 0, stream>>>(den, N * GAT_H, 0.0f);
    kepce_fill<<<kepce_cdiv(N * 16, B), B, 0, stream>>>(hB, N * 16, 0.0f);
    kepce_edge_alpha<<<kepce_cdiv(EN, B), B, 0, stream>>>(xl, xr, ei, ew, ce, lat, w1e, att1, alpha, amax, E, N, 4, 16);
    kepce_edge_expsum<<<kepce_cdiv(EN, B), B, 0, stream>>>(alpha, amax, den, ei, E, N);
    kepce_edge_aggr<<<kepce_cdiv(EN * GAT_H, B), B, 0, stream>>>(alpha, den, xl, ei, hB, E, N, 4, 16);
    kepce_bias_relu<<<kepce_cdiv(N * 16, B), B, 0, stream>>>(hB, bias1, N, 16);

    // ================= GAT layer 2: in 16, M=32, C=8 -> hA =================
    kepce_dual_linear<<<kepce_cdiv(N * 32, B), B, 0, stream>>>(hB, w2l, b2l, w2r, b2r, xl, xr, N, 16, 32);
    kepce_fill<<<kepce_cdiv(N * GAT_H, B), B, 0, stream>>>((float*)amax, N * GAT_H, 0.0f);
    kepce_fill<<<kepce_cdiv(N * GAT_H, B), B, 0, stream>>>(den, N * GAT_H, 0.0f);
    kepce_fill<<<kepce_cdiv(N * 32, B), B, 0, stream>>>(hA, N * 32, 0.0f);
    kepce_edge_alpha<<<kepce_cdiv(EN, B), B, 0, stream>>>(xl, xr, ei, ew, ce, lat, w2e, att2, alpha, amax, E, N, 8, 32);
    kepce_edge_expsum<<<kepce_cdiv(EN, B), B, 0, stream>>>(alpha, amax, den, ei, E, N);
    kepce_edge_aggr<<<kepce_cdiv(EN * GAT_H, B), B, 0, stream>>>(alpha, den, xl, ei, hA, E, N, 8, 32);
    kepce_bias_relu<<<kepce_cdiv(N * 32, B), B, 0, stream>>>(hA, bias2, N, 32);

    // --- FC 32->32 (relu) via WMMA, f16 output straight into hB region ---
    _Float16* h16 = (_Float16*)hB;   // h1 no longer needed
    {
        long tilesN = (N + 15) / 16;
        int blocksFC = kepce_cdiv(tilesN, 4);
        kepce_fc_wmma<<<blocksFC, 128, 0, stream>>>(hA, w_fc, b_fc, h16, N);
    }

    // --- edge MLP via WMMA ---
    {
        long tiles = (E + 15) / 16;
        int blocks = kepce_cdiv(tiles, 4);
        kepce_edge_mlp_wmma<<<blocks, 128, 0, stream>>>(h16, ei, ew, ce,
                                                        w_e1, b_e1, w_e2, b_e2,
                                                        (float*)d_out, E);
    }
}
